// FeedForward_15393162789391
// MI455X (gfx1250) — matrix operand order repaired and checked by reading, not compiled, not run on hardware
//
#include <hip/hip_runtime.h>
#include <math.h>

#define D_DIM 1024
#define F_DIM 4096
#define E_NUM 8
#define FC    256            // F-chunk per outer iteration
#define XP    (D_DIM + 8)    // LDS pitch (bf16 elems) for x tile  -> avoids bank conflicts, keeps 16B align
#define HP    (FC + 8)       // LDS pitch (bf16 elems) for h chunk

typedef __attribute__((ext_vector_type(16))) __bf16 bf16x16;
typedef __attribute__((ext_vector_type(8)))  __bf16 bf16x8;
typedef __attribute__((ext_vector_type(8)))  float  f32x8;

// ---- WMMA operand loaders (wave32, 16x16x32 bf16) -------------------------
// A (16x32, 16-bit): lane m=l&15, half=l>>4; elements 0..15 = K half*16+0..15 (matches B)
__device__ __forceinline__ bf16x16 load_a_lds(const __bf16* base, int pitch, int k0) {
    int lane = threadIdx.x & 31;
    int m = lane & 15, half = lane >> 4;
    const __bf16* p = base + m * pitch + k0 + half * 16;
    bf16x8 lo = *(const bf16x8*)p;
    bf16x8 hi = *(const bf16x8*)(p + 8);
    return __builtin_shufflevector(lo, hi, 0,1,2,3,4,5,6,7,8,9,10,11,12,13,14,15);
}
// B (32x16, 16-bit), weights pre-transposed N-major [n][k]: lane n=l&15 reads 16 contiguous K at half*16
__device__ __forceinline__ bf16x16 load_b_glb(const __bf16* base, int ktot, int n0, int k0) {
    int lane = threadIdx.x & 31;
    int n = lane & 15, half = lane >> 4;
    return *(const bf16x16*)(base + (size_t)(n0 + n) * ktot + k0 + half * 16);
}

__device__ __forceinline__ float gelu_tanh(float v) {
    return 0.5f * v * (1.0f + tanhf(0.7978845608028654f * (v + 0.044715f * v * v * v)));
}

// ---------------------------------------------------------------------------
__global__ void init_kernel(int* counts) {
    if (threadIdx.x < 2 * E_NUM) counts[threadIdx.x] = 0;
}

__global__ void convert_bf16_kernel(const float* __restrict__ in, __bf16* __restrict__ out, size_t n) {
    size_t i = (size_t)blockIdx.x * blockDim.x + threadIdx.x;
    if (i < n) out[i] = (__bf16)in[i];
}

// in: [z][R][C] f32  ->  out: [z][C][R] bf16   (tiled transpose + downconvert)
__global__ __launch_bounds__(256) void transpose_bf16_kernel(
    const float* __restrict__ in, __bf16* __restrict__ out, int R, int C)
{
    __shared__ float tile[32][33];
    const float* ine  = in  + (size_t)blockIdx.z * R * C;
    __bf16*      oute = out + (size_t)blockIdx.z * R * C;
    int c0 = blockIdx.x * 32, r0 = blockIdx.y * 32;
    int tx = threadIdx.x & 31, ty = threadIdx.x >> 5;   // 32 x 8
    #pragma unroll
    for (int j = 0; j < 32; j += 8)
        tile[ty + j][tx] = ine[(size_t)(r0 + ty + j) * C + c0 + tx];
    __syncthreads();
    #pragma unroll
    for (int j = 0; j < 32; j += 8)
        oute[(size_t)(c0 + ty + j) * R + r0 + tx] = (__bf16)tile[tx][ty + j];
}

// ---- Router: one wave per token -------------------------------------------
__global__ __launch_bounds__(256) void router_kernel(
    const float* __restrict__ x, const float* __restrict__ Wr,
    float* __restrict__ probs, float* __restrict__ slotw,
    int* __restrict__ lists, int* __restrict__ counts, int T)
{
    __shared__ float sWr[D_DIM * E_NUM];   // 32 KB
    int tid = threadIdx.x;
    for (int i = tid; i < D_DIM * E_NUM; i += 256) sWr[i] = Wr[i];
    __syncthreads();

    int wv = tid >> 5, lane = tid & 31;
    int t = blockIdx.x * 8 + wv;
    if (t >= T) return;

    float acc[E_NUM];
    #pragma unroll
    for (int e = 0; e < E_NUM; ++e) acc[e] = 0.0f;
    for (int d = lane; d < D_DIM; d += 32) {
        float xv = x[(size_t)t * D_DIM + d];
        #pragma unroll
        for (int e = 0; e < E_NUM; ++e) acc[e] += xv * sWr[d * E_NUM + e];
    }
    #pragma unroll
    for (int e = 0; e < E_NUM; ++e)
        #pragma unroll
        for (int off = 16; off >= 1; off >>= 1)
            acc[e] += __shfl_xor(acc[e], off, 32);

    // softmax over 8 (replicated per lane)
    float mx = acc[0];
    #pragma unroll
    for (int e = 1; e < E_NUM; ++e) mx = fmaxf(mx, acc[e]);
    float p[E_NUM], sum = 0.0f;
    #pragma unroll
    for (int e = 0; e < E_NUM; ++e) { p[e] = __expf(acc[e] - mx); sum += p[e]; }
    float inv = 1.0f / sum;
    #pragma unroll
    for (int e = 0; e < E_NUM; ++e) p[e] *= inv;

    if (lane == 0) {
        #pragma unroll
        for (int e = 0; e < E_NUM; ++e) probs[(size_t)t * E_NUM + e] = p[e];
        int i0 = 0;
        #pragma unroll
        for (int e = 1; e < E_NUM; ++e) if (p[e] > p[i0]) i0 = e;   // lower index wins ties
        int i1 = (i0 == 0) ? 1 : 0;
        #pragma unroll
        for (int e = 0; e < E_NUM; ++e) if (e != i0 && p[e] > p[i1]) i1 = e;
        float wsum = p[i0] + p[i1];
        slotw[t]     = p[i0] / wsum;
        slotw[T + t] = p[i1] / wsum;
        int pos0 = atomicAdd(&counts[i0 * 2 + 0], 1);
        lists[(size_t)(i0 * 2 + 0) * T + pos0] = t;
        int pos1 = atomicAdd(&counts[i1 * 2 + 1], 1);
        lists[(size_t)(i1 * 2 + 1) * T + pos1] = t;
    }
}

// ---- Fused expert FFN: 16-token tile x 8 waves, bf16 WMMA -----------------
__global__ __launch_bounds__(256) void ffn_kernel(
    const __bf16* __restrict__ xbf, const __bf16* __restrict__ w1t,
    const __bf16* __restrict__ w2t, const float* __restrict__ b1,
    const float* __restrict__ b2, const int* __restrict__ lists,
    const int* __restrict__ counts, const float* __restrict__ slotw,
    float* __restrict__ out, int T, int slot)
{
    __shared__ __align__(16) __bf16 xtile[16 * XP];  // 33.0 KB
    __shared__ __align__(16) __bf16 hch[16 * HP];    //  8.4 KB
    __shared__ int   stok[16];
    __shared__ float swt[16];

    const int e   = blockIdx.y;
    const int ls  = e * 2 + slot;
    const int cnt = counts[ls];
    const int row0 = blockIdx.x * 16;
    if (row0 >= cnt) return;                 // uniform exit: EXEC stays full for WMMA
    const int rows = (cnt - row0 < 16) ? (cnt - row0) : 16;

    const int tid  = threadIdx.x;
    const int wv   = tid >> 5;               // wave 0..7 -> output cols [wv*128, wv*128+128)
    const int lane = tid & 31;
    const int half = lane >> 4;

    if (tid < 16) {
        int tok = (tid < rows) ? lists[(size_t)ls * T + row0 + tid] : 0;
        stok[tid] = tok;
        swt[tid]  = (tid < rows) ? slotw[(size_t)slot * T + tok] : 0.0f;
    }
    __syncthreads();

    // stage x tile (16 x 1024 bf16); zero-pad invalid rows
    for (int i = tid; i < 16 * (D_DIM / 8); i += 256) {
        int m = i >> 7;                       // D/8 == 128
        int c = (i & 127) * 8;
        uint4 v = make_uint4(0, 0, 0, 0);
        if (m < rows) v = *(const uint4*)(xbf + (size_t)stok[m] * D_DIM + c);
        *(uint4*)(xtile + m * XP + c) = v;
    }
    __syncthreads();

    const __bf16* w1e = w1t + (size_t)e * F_DIM * D_DIM;   // [F][D]
    const __bf16* w2e = w2t + (size_t)e * D_DIM * F_DIM;   // [D][F]

    f32x8 zero = {};
    f32x8 acc[8];
    #pragma unroll
    for (int i = 0; i < 8; ++i) acc[i] = zero;

    for (int fc = 0; fc < F_DIM / FC; ++fc) {
        // GEMM1: this wave computes h[:, fbase .. fbase+32) = gelu(x @ W1 + b1)
        int fbase = fc * FC + wv * 32;
        #pragma unroll
        for (int nt = 0; nt < 2; ++nt) {
            f32x8 hacc = zero;
            int n0 = fbase + nt * 16;
            for (int k = 0; k < D_DIM; k += 32) {
                bf16x16 a = load_a_lds(xtile, XP, k);
                bf16x16 b = load_b_glb(w1e, D_DIM, n0, k);
                hacc = __builtin_amdgcn_wmma_f32_16x16x32_bf16(
                           false, a, false, b, (short)0, hacc, false, false);
            }
            int n = lane & 15;
            float bias = b1[e * F_DIM + n0 + n];
            int lcol = wv * 32 + nt * 16 + n;
            #pragma unroll
            for (int r = 0; r < 8; ++r) {
                int m = r + half * 8;
                hch[m * HP + lcol] = (__bf16)gelu_tanh(hacc[r] + bias);
            }
        }
        __syncthreads();
        // GEMM2: acc += h_chunk @ W2[chunk, wv*128 .. +128)
        for (int k2 = 0; k2 < FC; k2 += 32) {
            bf16x16 a2 = load_a_lds(hch, HP, k2);
            int kg = fc * FC + k2;
            #pragma unroll
            for (int nt2 = 0; nt2 < 8; ++nt2) {
                bf16x16 bv = load_b_glb(w2e, F_DIM, wv * 128 + nt2 * 16, kg);
                acc[nt2] = __builtin_amdgcn_wmma_f32_16x16x32_bf16(
                               false, a2, false, bv, (short)0, acc[nt2], false, false);
            }
        }
        __syncthreads();                      // protect hch before next chunk overwrites it
    }

    // epilogue: out[t, d] (slot0: store w*(y+b2); slot1: accumulate)
    int n = lane & 15;
    #pragma unroll
    for (int nt2 = 0; nt2 < 8; ++nt2) {
        int d = wv * 128 + nt2 * 16 + n;
        float bb = b2[e * D_DIM + d];
        #pragma unroll
        for (int r = 0; r < 8; ++r) {
            int m = r + half * 8;
            if (m < rows) {
                float v = (acc[nt2][r] + bb) * swt[m];
                float* dst = out + (size_t)stok[m] * D_DIM + d;
                if (slot == 0) *dst = v; else *dst += v;
            }
        }
    }
}

// ---- Aux loss: fixed-order reduction (deterministic) ----------------------
__global__ __launch_bounds__(256) void loss_kernel(
    const float* __restrict__ probs, const int* __restrict__ counts,
    float* __restrict__ out_loss, int T)
{
    __shared__ float red[256 * E_NUM];
    int tid = threadIdx.x;
    float s[E_NUM];
    #pragma unroll
    for (int e = 0; e < E_NUM; ++e) s[e] = 0.0f;
    for (int t = tid; t < T; t += 256)
        #pragma unroll
        for (int e = 0; e < E_NUM; ++e) s[e] += probs[(size_t)t * E_NUM + e];
    #pragma unroll
    for (int e = 0; e < E_NUM; ++e) red[tid * E_NUM + e] = s[e];
    __syncthreads();
    for (int stride = 128; stride > 0; stride >>= 1) {
        if (tid < stride)
            #pragma unroll
            for (int e = 0; e < E_NUM; ++e)
                red[tid * E_NUM + e] += red[(tid + stride) * E_NUM + e];
        __syncthreads();
    }
    if (tid == 0) {
        float invT = 1.0f / (float)T, loss = 0.0f;
        for (int e = 0; e < E_NUM; ++e) {
            float meanp = red[e] * invT;
            float frac  = (float)(counts[e * 2] + counts[e * 2 + 1]) * invT;
            loss += frac * meanp;
        }
        *out_loss = (float)E_NUM * loss;
    }
}

// ---------------------------------------------------------------------------
extern "C" void kernel_launch(void* const* d_in, const int* in_sizes, int n_in,
                              void* d_out, int out_size, void* d_ws, size_t ws_size,
                              hipStream_t stream)
{
    (void)n_in; (void)out_size; (void)ws_size;
    const float* x  = (const float*)d_in[0];
    const float* Wr = (const float*)d_in[1];
    const float* W1 = (const float*)d_in[2];
    const float* b1 = (const float*)d_in[3];
    const float* W2 = (const float*)d_in[4];
    const float* b2 = (const float*)d_in[5];
    float* out = (float*)d_out;
    const int T = in_sizes[0] / D_DIM;       // 4096

    char* ws = (char*)d_ws;
    size_t off = 0;
    auto alloc = [&](size_t bytes) { void* p = ws + off; off += (bytes + 255) & ~(size_t)255; return p; };
    __bf16* xbf   = (__bf16*)alloc((size_t)T * D_DIM * 2);                 //   8 MB
    __bf16* w1t   = (__bf16*)alloc((size_t)E_NUM * D_DIM * F_DIM * 2);     //  64 MB  [E][F][D]
    __bf16* w2t   = (__bf16*)alloc((size_t)E_NUM * D_DIM * F_DIM * 2);     //  64 MB  [E][D][F]
    float*  probs = (float*)alloc((size_t)T * E_NUM * 4);
    float*  slotw = (float*)alloc((size_t)2 * T * 4);
    int*    lists = (int*)alloc((size_t)2 * E_NUM * T * 4);
    int*    counts= (int*)alloc(256);

    init_kernel<<<1, 32, 0, stream>>>(counts);

    size_t nx = (size_t)T * D_DIM;
    convert_bf16_kernel<<<(int)((nx + 255) / 256), 256, 0, stream>>>(x, xbf, nx);
    // W1 [E][D][F] -> [E][F][D] bf16 ; W2 [E][F][D] -> [E][D][F] bf16
    transpose_bf16_kernel<<<dim3(F_DIM / 32, D_DIM / 32, E_NUM), 256, 0, stream>>>(W1, w1t, D_DIM, F_DIM);
    transpose_bf16_kernel<<<dim3(D_DIM / 32, F_DIM / 32, E_NUM), 256, 0, stream>>>(W2, w2t, F_DIM, D_DIM);

    router_kernel<<<T / 8, 256, 0, stream>>>(x, Wr, probs, slotw, lists, counts, T);

    dim3 g(T / 16, E_NUM);
    ffn_kernel<<<g, 256, 0, stream>>>(xbf, w1t, w2t, b1, b2, lists, counts, slotw, out, T, 0);
    ffn_kernel<<<g, 256, 0, stream>>>(xbf, w1t, w2t, b1, b2, lists, counts, slotw, out, T, 1);

    loss_kernel<<<1, 256, 0, stream>>>(probs, counts, out + (size_t)T * D_DIM, T);
}
